// MultiHeadAttention_61383672594468
// MI455X (gfx1250) — matrix operand order repaired and checked by reading, not compiled, not run on hardware
//
#include <hip/hip_runtime.h>
#include <hip/hip_bf16.h>

// ---------------------------------------------------------------------------
// MultiHeadAttention (BN -> QKV proj -> softmax(QK^T/8)V -> out proj) on
// gfx1250: v_wmma_f32_16x16x32_f16 for every matmul, TDM (tensor_load_to_lds)
// for GEMM tile staging, async global->LDS copies for attention K tiles.
// B=8, S=1024, C=INNER=512, H=8, D=64.
// ---------------------------------------------------------------------------

typedef __attribute__((ext_vector_type(16))) _Float16 v16h;
typedef __attribute__((ext_vector_type(8)))  float    v8f;
typedef __attribute__((ext_vector_type(4)))  unsigned int u32x4;
typedef __attribute__((ext_vector_type(4)))  unsigned int v4u;
typedef __attribute__((ext_vector_type(8)))  int          v8i;
typedef __attribute__((ext_vector_type(4)))  int          v4i;

union AFrag { unsigned int u[8]; v16h v; };   // A-matrix 16x32 f16 fragment
union BFrag { u32x4 q[2];        v16h v; };   // B-matrix 32x16 f16 fragment

#define NB  8
#define SS  1024
#define CC  512
#define HH  8
#define DD  64

// workspace layout (bytes)
#define OFF_T   ((size_t)0)                         // t16 [B*S, C] f16   8 MB
#define OFF_WQ  ((size_t)8388608)                   // wq16 [512,512] f16
#define OFF_WK  (OFF_WQ + 524288)
#define OFF_WV  (OFF_WK + 524288)
#define OFF_WO  (OFF_WV + 524288)
#define OFF_Q   (OFF_WO + 524288)                   // Q16 [B,H,S,D] f16  8 MB
#define OFF_K   (OFF_Q + 8388608)
#define OFF_V   (OFF_K + 8388608)
#define OFF_O   (OFF_V + 8388608)                   // O16 [B*S, 512] f16 8 MB

#ifndef __has_builtin
#define __has_builtin(x) 0
#endif
#if defined(__AMDGCN__) && __has_builtin(__builtin_amdgcn_tensor_load_to_lds)
#define HAVE_TDM 1
#else
#define HAVE_TDM 0
#endif

#if HAVE_TDM
// ---------------------------------------------------------------------------
// Issue one TDM 2D tile load (global -> LDS). Descriptor bitfields per CDNA5
// ISA 8.3/8.4: group0 = {count, lds_addr, global_addr[56:0], type=2},
// group1 = {wg_mask=0, data_size=1 (2B), tensor_dim0/1, tile_dim0/1,
//           tensor_dim0_stride}. 2D only: groups 2/3 zero, tile_dim2=0.
// All arguments are workgroup-uniform. Tracked by TENSORcnt.
// ---------------------------------------------------------------------------
__device__ __forceinline__ void tdm_load_2d_f16(unsigned lds_byte_addr,
                                                unsigned long long gaddr,
                                                unsigned tile0_elems,
                                                unsigned tile1_rows,
                                                unsigned tdim0_elems,
                                                unsigned tdim1_rows,
                                                unsigned long long row_stride_elems) {
    v4u g0;
    g0[0] = 1u;                                        // count=1 valid descriptor
    g0[1] = lds_byte_addr;                             // lds_addr
    g0[2] = (unsigned)(gaddr & 0xFFFFFFFFull);         // global_addr[31:0]
    g0[3] = (unsigned)((gaddr >> 32) & 0x1FFFFFFu)     // global_addr[56:32]
          | (2u << 30);                                // type=2 ("image")
    v8i g1;
    g1[0] = (int)(1u << 16);                           // data_size=1 -> 2 bytes
    g1[1] = (int)((tdim0_elems & 0xFFFFu) << 16);      // tensor_dim0[15:0]
    g1[2] = (int)(((tdim0_elems >> 16) & 0xFFFFu)
          | ((tdim1_rows & 0xFFFFu) << 16));           // dim0[31:16] | dim1[15:0]
    g1[3] = (int)(((tdim1_rows >> 16) & 0xFFFFu)
          | ((tile0_elems & 0xFFFFu) << 16));          // dim1[31:16] | tile_dim0
    g1[4] = (int)(tile1_rows & 0xFFFFu);               // tile_dim1; tile_dim2=0
    g1[5] = (int)(unsigned)(row_stride_elems & 0xFFFFFFFFull);   // stride0[31:0]
    g1[6] = (int)(unsigned)((row_stride_elems >> 32) & 0xFFFFu); // stride0[47:32]
    g1[7] = 0;                                          // tensor_dim1_stride unused (2D)
    v4i gz = {0, 0, 0, 0};
#if __clang_major__ >= 23
    v8i gz8 = {0, 0, 0, 0, 0, 0, 0, 0};
    __builtin_amdgcn_tensor_load_to_lds(g0, g1, gz, gz, gz8, 0);
#else
    __builtin_amdgcn_tensor_load_to_lds(g0, g1, gz, gz, 0);
#endif
}
#endif  // HAVE_TDM

// ---------------------------------------------------------------------------
// float -> f16 weight pack (layout preserved: w16[n*512+k] = w[n][k], which is
// exactly the B-fragment friendly "row n holds K-contiguous" layout).
// ---------------------------------------------------------------------------
__global__ void cvt_w_kernel(const float* __restrict__ w, _Float16* __restrict__ o, int n) {
    int i = blockIdx.x * blockDim.x + threadIdx.x;
    if (i < n) o[i] = (_Float16)w[i];
}

// ---------------------------------------------------------------------------
// BatchNorm (eval) + transpose [B,C,S] -> [B*S, C], packed f16.
// ---------------------------------------------------------------------------
__global__ void bn_pack_kernel(const float* __restrict__ x,
                               const float* __restrict__ gamma,
                               const float* __restrict__ beta,
                               const float* __restrict__ mean,
                               const float* __restrict__ var,
                               _Float16* __restrict__ t16) {
    int tid = blockIdx.x * blockDim.x + threadIdx.x;     // over 8192*512
    int c   = tid & (CC - 1);
    int row = tid >> 9;                                  // b*S + s
    int s   = row & (SS - 1);
    int b   = row >> 10;
    float g   = gamma[c] * rsqrtf(var[c] + 1e-5f);
    float val = (x[((size_t)(b * CC + c) << 10) + s] - mean[c]) * g + beta[c];
    t16[tid] = (_Float16)val;
}

// ---------------------------------------------------------------------------
// WMMA GEMM:  out[m, n] = sum_k A[m,k] * W[n,k] + bias[n]
// Block = 128 threads (4 waves), block tile M=64 x N=64, K staged 32-wide in
// LDS by the Tensor Data Mover (one descriptor per matrix per k-step).
// Wave w owns N-subtile w*16 and all four 16-row M-subtiles.
// MODE 0: out f16 scattered to [B,H,S,D].   MODE 1: out f32 to y[b,c,s].
// ---------------------------------------------------------------------------
template <int MODE>
__global__ void __launch_bounds__(128) gemm_kernel(const _Float16* __restrict__ A,
                                                   const _Float16* __restrict__ W,
                                                   const float* __restrict__ bias,
                                                   void* __restrict__ out) {
    __shared__ __align__(16) _Float16 As[64 * 32];
    __shared__ __align__(16) _Float16 Bs[64 * 32];

    const int tid  = threadIdx.x;
    const int wave = tid >> 5;
    const int lane = tid & 31;
    const int l16  = lane & 15;
    const int grp  = lane >> 4;
    const int m0   = blockIdx.x * 64;
    const int n0   = blockIdx.y * 64;

    v8f acc[4];
#pragma unroll
    for (int i = 0; i < 4; ++i)
#pragma unroll
        for (int j = 0; j < 8; ++j) acc[i][j] = 0.0f;

#if HAVE_TDM
    const unsigned ldsA = (unsigned)(size_t)(void*)As;   // low 32b = LDS offset
    const unsigned ldsB = (unsigned)(size_t)(void*)Bs;
#endif

    for (int kt = 0; kt < CC; kt += 32) {
#if HAVE_TDM
        // ---- TDM staging: two 64x32-f16 tile DMAs, issued by wave 0 ----
        if (wave == 0) {
            unsigned long long gaA =
                (unsigned long long)(size_t)(A + (size_t)m0 * CC + kt);
            unsigned long long gaB =
                (unsigned long long)(size_t)(W + (size_t)n0 * CC + kt);
            tdm_load_2d_f16(ldsA, gaA, 32u, 64u, (unsigned)CC, (unsigned)(NB * SS),
                            (unsigned long long)CC);
            tdm_load_2d_f16(ldsB, gaB, 32u, 64u, (unsigned)CC, (unsigned)CC,
                            (unsigned long long)CC);
            __builtin_amdgcn_s_wait_tensorcnt((unsigned short)0);
        }
#else
        // ---- fallback: cooperative staging, 64 rows x 32 halves each ----
        u32x4* As4 = (u32x4*)As;
        u32x4* Bs4 = (u32x4*)Bs;
#pragma unroll
        for (int j = 0; j < 2; ++j) {
            int lin = j * 128 + tid;             // 0..255
            int r   = lin >> 2;
            int seg = lin & 3;
            As4[lin] = *(const u32x4*)(A + (size_t)(m0 + r) * CC + kt + seg * 8);
            Bs4[lin] = *(const u32x4*)(W + (size_t)(n0 + r) * CC + kt + seg * 8);
        }
#endif
        if (kt + 32 < CC) {                       // hint next K slab into cache
            __builtin_prefetch(A + (size_t)(m0 + (tid >> 1)) * CC + kt + 32, 0, 0);
            __builtin_prefetch(W + (size_t)(n0 + (tid >> 1)) * CC + kt + 32, 0, 0);
        }
        __syncthreads();

        // ---- B fragment for this wave's 16 columns ----
        BFrag bf;
        {
            int bn = wave * 16 + l16;
            int kb = grp * 16;                    // halves
            const u32x4* p = (const u32x4*)(Bs + bn * 32 + kb);
            bf.q[0] = p[0];
            bf.q[1] = p[1];
        }

        const unsigned int* AsU = (const unsigned int*)As;
#pragma unroll
        for (int mt = 0; mt < 4; ++mt) {
            AFrag af;
            int am = mt * 16 + l16;
#pragma unroll
            for (int v = 0; v < 8; ++v) {
                int kk  = 2 * v + grp * 16;
                af.u[v] = AsU[am * 16 + (kk >> 1)];
            }
            acc[mt] = __builtin_amdgcn_wmma_f32_16x16x32_f16(
                false, af.v, false, bf.v, (short)0, acc[mt], false, false);
        }
        __syncthreads();
    }

    // ---- epilogue: bias + scatter ----
    int   gn = n0 + wave * 16 + l16;
    float bv = bias[gn];
#pragma unroll
    for (int mt = 0; mt < 4; ++mt) {
#pragma unroll
        for (int r = 0; r < 8; ++r) {
            int   gm  = m0 + mt * 16 + r + grp * 8;   // global row = b*S + s
            float val = acc[mt][r] + bv;
            int b = gm >> 10, s = gm & (SS - 1);
            if (MODE == 0) {
                int h = gn >> 6, d = gn & 63;
                ((_Float16*)out)[(((size_t)(b * HH + h) * SS + s) << 6) + d] = (_Float16)val;
            } else {
                ((float*)out)[((size_t)(b * CC + gn) << 10) + s] = val;
            }
        }
    }
}

// ---------------------------------------------------------------------------
// Flash attention per (b, h, 64-query tile). 4 waves/block, 16 queries each.
// Online softmax over 32-key tiles; Q*K^T and P*V via WMMA. K tile staged
// with async global->LDS (ASYNCcnt), V staged manually (transposed).
// ---------------------------------------------------------------------------
__global__ void __launch_bounds__(128) attn_kernel(const _Float16* __restrict__ Q,
                                                   const _Float16* __restrict__ K,
                                                   const _Float16* __restrict__ V,
                                                   _Float16* __restrict__ O) {
    __shared__ __align__(16) _Float16 Ks[32 * 64];     // [key][d]
    __shared__ __align__(16) _Float16 Vt[64 * 32];     // [d][key]
    __shared__ __align__(16) _Float16 Ps[4][16 * 32];  // per-wave P tile

    const int tid  = threadIdx.x;
    const int wave = tid >> 5;
    const int lane = tid & 31;
    const int l16  = lane & 15;
    const int grp  = lane >> 4;

    const int bid = blockIdx.x;        // b*128 + h*16 + qt
    const int qt  = bid & 15;
    const int h   = (bid >> 4) & 7;
    const int b   = bid >> 7;
    const int s0w = qt * 64 + wave * 16;

    const _Float16* Qb = Q + ((size_t)(b * HH + h) << 16);   // *S*D
    const _Float16* Kb = K + ((size_t)(b * HH + h) << 16);
    const _Float16* Vb = V + ((size_t)(b * HH + h) << 16);

    // Q A-fragments (d = 0..31 and 32..63), loaded once
    AFrag qf[2];
    {
        const unsigned int* Qu = (const unsigned int*)Qb;
        int qm = s0w + l16;
#pragma unroll
        for (int c = 0; c < 2; ++c)
#pragma unroll
            for (int v = 0; v < 8; ++v) {
                int kk     = 2 * v + grp * 16 + c * 32;
                qf[c].u[v] = Qu[qm * 32 + (kk >> 1)];
            }
    }

    float Mr[8], Lr[8];
#pragma unroll
    for (int r = 0; r < 8; ++r) { Mr[r] = -1e30f; Lr[r] = 0.0f; }
    v8f oacc[4];
#pragma unroll
    for (int i = 0; i < 4; ++i)
#pragma unroll
        for (int j = 0; j < 8; ++j) oacc[i][j] = 0.0f;

    const float    scale = 0.125f;                       // 1/sqrt(64)
    const unsigned ldsK  = (unsigned)(size_t)(void*)Ks;  // low 32b = LDS offset

    for (int kt = 0; kt < SS; kt += 32) {
        __syncthreads();   // protect LDS reuse across iterations
        {
            // stage K tile [key][d] via async global->LDS b128 copies
            const char* gk = (const char*)(Kb + (size_t)kt * DD);
#pragma unroll
            for (int j = 0; j < 2; ++j) {
                int      lin = j * 128 + tid;          // 256 x 16B = 32x64 halves
                unsigned dst = ldsK + (unsigned)(lin * 16);
                const char* src = gk + (size_t)lin * 16;
                asm volatile("global_load_async_to_lds_b128 %0, %1, off"
                             :: "v"(dst), "v"(src) : "memory");
            }
            // stage V transposed: Vt[d][key]
            const unsigned int* Vg = (const unsigned int*)(Vb + (size_t)kt * DD);
#pragma unroll
            for (int j = 0; j < 8; ++j) {
                int lin = j * 128 + tid;               // 0..1023
                int key = lin >> 5;
                int dp  = lin & 31;
                union { unsigned int u; _Float16 hv[2]; } cv;
                cv.u = Vg[key * 32 + dp];
                Vt[(2 * dp) * 32 + key]     = cv.hv[0];
                Vt[(2 * dp + 1) * 32 + key] = cv.hv[1];
            }
            asm volatile("s_wait_asynccnt 0" ::: "memory");
        }
        __syncthreads();

        // ---- scores S = Q * K^T for 32 keys (2 N-subtiles, K-dim = 64) ----
        v8f sacc[2];
#pragma unroll
        for (int i = 0; i < 2; ++i)
#pragma unroll
            for (int j = 0; j < 8; ++j) sacc[i][j] = 0.0f;
#pragma unroll
        for (int nt = 0; nt < 2; ++nt) {
#pragma unroll
            for (int c = 0; c < 2; ++c) {
                BFrag kf;
                int keyloc = nt * 16 + l16;
                int kb     = grp * 16 + c * 32;      // d offset in halves
                const u32x4* p = (const u32x4*)(Ks + keyloc * 64 + kb);
                kf.q[0] = p[0];
                kf.q[1] = p[1];
                sacc[nt] = __builtin_amdgcn_wmma_f32_16x16x32_f16(
                    false, qf[c].v, false, kf.v, (short)0, sacc[nt], false, false);
            }
        }

        // ---- online softmax update (row reductions via wave32 shuffles) ----
        float pr[2][8];
        float alpha[8];
#pragma unroll
        for (int r = 0; r < 8; ++r) {
            float sv0 = sacc[0][r] * scale;
            float sv1 = sacc[1][r] * scale;
            float mx  = fmaxf(sv0, sv1);
#pragma unroll
            for (int off = 1; off < 16; off <<= 1)
                mx = fmaxf(mx, __shfl_xor(mx, off, 32));
            float Mn = fmaxf(Mr[r], mx);
            float a  = __expf(Mr[r] - Mn);
            float p0 = __expf(sv0 - Mn);
            float p1 = __expf(sv1 - Mn);
            float rs = p0 + p1;
#pragma unroll
            for (int off = 1; off < 16; off <<= 1)
                rs += __shfl_xor(rs, off, 32);
            Lr[r]    = Lr[r] * a + rs;
            Mr[r]    = Mn;
            alpha[r] = a;
            pr[0][r] = p0;
            pr[1][r] = p1;
        }
#pragma unroll
        for (int i = 0; i < 4; ++i)
#pragma unroll
            for (int r = 0; r < 8; ++r) oacc[i][r] *= alpha[r];

        // ---- C-layout -> A-fragment layout for P via per-wave LDS tile ----
        _Float16* Pw = Ps[wave];
#pragma unroll
        for (int nt = 0; nt < 2; ++nt)
#pragma unroll
            for (int r = 0; r < 8; ++r) {
                int m = r + grp * 8;
                int n = nt * 16 + l16;
                Pw[m * 32 + n] = (_Float16)pr[nt][r];
            }
        asm volatile("s_wait_dscnt 0" ::: "memory");   // intra-wave LDS RAW

        AFrag pf;
        {
            const unsigned int* Pu = (const unsigned int*)Pw;
#pragma unroll
            for (int v = 0; v < 8; ++v) {
                int kk  = 2 * v + grp * 16;
                pf.u[v] = Pu[l16 * 16 + (kk >> 1)];
            }
        }

        // ---- O += P * V (4 N-subtiles over d) ----
#pragma unroll
        for (int nt = 0; nt < 4; ++nt) {
            BFrag vf;
            int dn = nt * 16 + l16;
            int kb = grp * 16;                    // key offset in halves
            const u32x4* p = (const u32x4*)(Vt + dn * 32 + kb);
            vf.q[0]  = p[0];
            vf.q[1]  = p[1];
            oacc[nt] = __builtin_amdgcn_wmma_f32_16x16x32_f16(
                false, pf.v, false, vf.v, (short)0, oacc[nt], false, false);
        }
    }

    // ---- normalize and write O16 [B*S, 512] ----
#pragma unroll
    for (int r = 0; r < 8; ++r) {
        float  inv = 1.0f / Lr[r];
        int    s   = s0w + r + grp * 8;
        size_t row = (size_t)b * SS + s;
#pragma unroll
        for (int nt = 0; nt < 4; ++nt) {
            int col = h * DD + nt * 16 + l16;
            O[row * CC + col] = (_Float16)(oacc[nt][r] * inv);
        }
    }
}

// ---------------------------------------------------------------------------
extern "C" void kernel_launch(void* const* d_in, const int* in_sizes, int n_in,
                              void* d_out, int out_size, void* d_ws, size_t ws_size,
                              hipStream_t stream) {
    const float* x     = (const float*)d_in[0];
    const float* gamma = (const float*)d_in[1];
    const float* beta  = (const float*)d_in[2];
    const float* mean  = (const float*)d_in[3];
    const float* var   = (const float*)d_in[4];
    const float* wq    = (const float*)d_in[5];
    const float* bq    = (const float*)d_in[6];
    const float* wk    = (const float*)d_in[7];
    const float* bk    = (const float*)d_in[8];
    const float* wv    = (const float*)d_in[9];
    const float* bv    = (const float*)d_in[10];
    const float* wo    = (const float*)d_in[11];
    const float* bo    = (const float*)d_in[12];

    char* ws = (char*)d_ws;
    _Float16* t16  = (_Float16*)(ws + OFF_T);
    _Float16* wq16 = (_Float16*)(ws + OFF_WQ);
    _Float16* wk16 = (_Float16*)(ws + OFF_WK);
    _Float16* wv16 = (_Float16*)(ws + OFF_WV);
    _Float16* wo16 = (_Float16*)(ws + OFF_WO);
    _Float16* Q16  = (_Float16*)(ws + OFF_Q);
    _Float16* K16  = (_Float16*)(ws + OFF_K);
    _Float16* V16  = (_Float16*)(ws + OFF_V);
    _Float16* O16  = (_Float16*)(ws + OFF_O);

    const int WN = CC * CC;   // 262144
    cvt_w_kernel<<<WN / 256, 256, 0, stream>>>(wq, wq16, WN);
    cvt_w_kernel<<<WN / 256, 256, 0, stream>>>(wk, wk16, WN);
    cvt_w_kernel<<<WN / 256, 256, 0, stream>>>(wv, wv16, WN);
    cvt_w_kernel<<<WN / 256, 256, 0, stream>>>(wo, wo16, WN);

    bn_pack_kernel<<<(NB * SS * CC) / 256, 256, 0, stream>>>(x, gamma, beta, mean, var, t16);

    dim3 ggrid(NB * SS / 64, CC / 64);   // (128, 8)
    gemm_kernel<0><<<ggrid, 128, 0, stream>>>(t16, wq16, bq, (void*)Q16);
    gemm_kernel<0><<<ggrid, 128, 0, stream>>>(t16, wk16, bk, (void*)K16);
    gemm_kernel<0><<<ggrid, 128, 0, stream>>>(t16, wv16, bv, (void*)V16);

    attn_kernel<<<NB * HH * (SS / 64), 128, 0, stream>>>(Q16, K16, V16, O16);

    gemm_kernel<1><<<ggrid, 128, 0, stream>>>(O16, wo16, bo, d_out);
}
